// STGNN_52115133170279
// MI455X (gfx1250) — matrix operand order repaired and checked by reading, not compiled, not run on hardware
//
#include <hip/hip_runtime.h>
#include <hip/hip_bf16.h>

// ---------------------------------------------------------------------------
// STGNN forward for MI455X (gfx1250, wave32).  All heavy GEMMs run on
// v_wmma_f32_16x16x32_f16.  Weights are converted f32->f16 and pre-swizzled
// into WMMA B-fragment order in d_ws each call; activations stay in LDS inside
// fused per-sequence kernels.  Fragment traffic is vectorized to b128.
//
// Input index map (setup_inputs dict order, params pytree flattened in dict /
// list insertion order):
//  0 positions  1 goals  2 history  3 local_maps  4 adjacency  5 edge_features
//  6 pos_W 7 pos_b 8 goal_W 9 goal_b
//  10..25 hist_layers[0]{Wq,Wk,Wv,Wo,bq,bk,bv,bo,ln1_g,ln1_b,W1,b1,W2,b2,ln2_g,ln2_b}
//  26..41 hist_layers[1]
//  42 conv1_W 43 conv1_b 44 conv2_W 45 conv2_b
//  46..61 temp_layers[0]  62..77 temp_layers[1]
//  78..89 spat_layers[0]{Wq,Wk,Wv,Wo,bq,bk,bv,bo,We,be,ln_g,ln_b}  90..101 spat_layers[1]
//  102 out_W 103 out_b
// ---------------------------------------------------------------------------

typedef __attribute__((ext_vector_type(16))) _Float16 v16h;
typedef __attribute__((ext_vector_type(8)))  _Float16 v8h;
typedef __attribute__((ext_vector_type(8)))  float    v8f;

#define NB  4
#define NT  8
#define NN  64
#define NL  16
#define ND  128
#define NE  16
#define NP  12
#define NOUT 2
static __device__ __host__ constexpr float SCALE_QK = 0.17677669529663687f; // 1/sqrt(32)

// ------------------------- WMMA fragment helpers ---------------------------
__device__ inline v8f wmma_f16(v16h a, v16h b, v8f c) {
  return __builtin_amdgcn_wmma_f32_16x16x32_f16(false, a, false, b, (short)0, c,
                                                false, false);
}

__device__ inline v16h join8(v8h lo, v8h hi) {
  return __builtin_shufflevector(lo, hi, 0, 1, 2, 3, 4, 5, 6, 7, 8, 9, 10, 11,
                                 12, 13, 14, 15);
}

// A fragment: 16x32 f16 tile (rows 0..15, cols k0..k0+31) of row-major src.
// Two aligned 16B runs per lane -> ds_load_b128 pairs.
__device__ inline v16h load_a(const _Float16* src, int ld, int k0) {
  const int lane = threadIdx.x & 31;
  const int m = lane & 15, hi = lane >> 4;
  const _Float16* p = src + m * ld + k0 + hi * 16;
  return join8(*(const v8h*)p, *(const v8h*)(p + 8));
}

// B fragment from packed (fragment-order) weights: tile (kt,nt) of a matrix
// with ntt = N/16 column tiles.  One contiguous 32B run per lane.
__device__ inline v16h load_bp(const _Float16* pk, int ntt, int kt, int nt) {
  const int lane = threadIdx.x & 31;
  const _Float16* p = pk + ((((size_t)kt * ntt + nt) * 32 + lane) << 4);
  return join8(*(const v8h*)p, *(const v8h*)(p + 8));
}

// B fragment from a transposed LDS view: B[k][n] = src[n0+n][colbase+k0+k]
// (used for Q @ K^T).  Contiguous 16-half run per lane.
__device__ inline v16h load_bT(const _Float16* src, int ld, int colbase, int k0,
                               int n0) {
  const int lane = threadIdx.x & 31;
  const int n = n0 + (lane & 15);
  const int k = k0 + ((lane >> 4) << 4);
  const _Float16* p = src + (size_t)n * ld + colbase + k;
  return join8(*(const v8h*)p, *(const v8h*)(p + 8));
}

// ----------------------------- weight structs ------------------------------
struct TLayerW {
  const _Float16 *Wq, *Wk, *Wv, *Wo, *W1, *W2;  // packed B-fragment order
  const float *bq, *bk, *bv, *bo, *b1, *b2, *g1, *be1, *g2, *be2;
};
struct SLayerW {
  const _Float16 *Wq, *Wk, *Wv, *Wo;
  const float *bq, *bk, *bv, *bo, *We, *be, *g, *bln;
};

// ------------------------ shared-memory footprints -------------------------
struct SmemT {                 // 49,152 B
  float    xF[16][128];
  _Float16 aH[16][128];
  _Float16 qH[16][128];
  _Float16 kH[16][128];
  _Float16 vT[128][32];        // V transposed: vT[d][token], tokens 16..31 = 0
  float    sc[4][16][16];
  _Float16 attH[4][16][32];
  _Float16 oH[16][128];
  _Float16 pH[16][256];
};

// post-LN over 128 cols, 16 rows, 8 waves (2 rows/wave), lane covers 4 cols.
__device__ inline void layernorm16(float (*xF)[128], const float* g,
                                   const float* b) {
  const int wave = threadIdx.x >> 5, lane = threadIdx.x & 31;
#pragma unroll
  for (int rr = 0; rr < 2; ++rr) {
    const int row = wave * 2 + rr;
    float v[4], s = 0.f, s2 = 0.f;
#pragma unroll
    for (int i = 0; i < 4; ++i) {
      v[i] = xF[row][lane * 4 + i];
      s += v[i];
      s2 += v[i] * v[i];
    }
#pragma unroll
    for (int m = 16; m >= 1; m >>= 1) {
      s += __shfl_xor(s, m);
      s2 += __shfl_xor(s2, m);
    }
    const float mean = s * (1.f / 128.f);
    const float var = s2 * (1.f / 128.f) - mean * mean;
    const float rstd = rsqrtf(var + 1e-5f);
#pragma unroll
    for (int i = 0; i < 4; ++i) {
      const int col = lane * 4 + i;
      xF[row][col] = (v[i] - mean) * rstd * g[col] + b[col];
    }
  }
}

// cast the 16x128 f32 activation block to f16 with one b128 store per thread.
__device__ inline void cast_act16(SmemT& s) {
  const int tid = threadIdx.x;
  const float* xf = &s.xF[0][0] + tid * 8;
  v8h hv;
#pragma unroll
  for (int j = 0; j < 8; ++j) hv[j] = (_Float16)xf[j];
  *(v8h*)(&s.aH[0][0] + tid * 8) = hv;
}

// -------------- fused post-LN transformer encoder layer (S<=16) ------------
template <int SREAL, int FDIM>
__device__ void tblock(SmemT& s, const TLayerW& w) {
  const int tid = threadIdx.x;
  const int wave = tid >> 5, lane = tid & 31;
  const int ln15 = lane & 15, hi = lane >> 4;
  const int n0 = wave * 16;
  constexpr int NCHUNK = FDIM / 256;

  cast_act16(s);
  __syncthreads();

  // ---- Q,K projections (row-major) and V projection (transposed store)
  {
    const _Float16* Wm[2] = {w.Wq, w.Wk};
    const float* bm[2] = {w.bq, w.bk};
    _Float16(*dst[2])[128] = {s.qH, s.kH};
#pragma unroll
    for (int which = 0; which < 2; ++which) {
      v8f c = {};
#pragma unroll
      for (int k0 = 0; k0 < 128; k0 += 32) {
        v16h a = load_a(&s.aH[0][0], 128, k0);
        v16h b = load_bp(Wm[which], 8, k0 >> 5, wave);
        c = wmma_f16(a, b, c);
      }
      const float bias = bm[which][n0 + ln15];
#pragma unroll
      for (int r = 0; r < 8; ++r)
        dst[which][r + 8 * hi][n0 + ln15] = (_Float16)(c[r] + bias);
    }
    {
      v8f c = {};
#pragma unroll
      for (int k0 = 0; k0 < 128; k0 += 32) {
        v16h a = load_a(&s.aH[0][0], 128, k0);
        v16h b = load_bp(w.Wv, 8, k0 >> 5, wave);
        c = wmma_f16(a, b, c);
      }
      const float bias = w.bv[n0 + ln15];
      v8h hv, z = {};
#pragma unroll
      for (int r = 0; r < 8; ++r) hv[r] = (_Float16)(c[r] + bias);
      *(v8h*)&s.vT[n0 + ln15][8 * hi] = hv;        // tokens 0..15
      *(v8h*)&s.vT[n0 + ln15][16 + 8 * hi] = z;    // zero pad tokens 16..31
    }
  }
  __syncthreads();

  // ---- scores per head: S_h = Q_h (16x32) @ K_h^T (32x16)
  if (wave < 4) {
    const int hd = wave;
    v16h a = load_a(&s.qH[0][0], 128, 32 * hd);
    v16h b = load_bT(&s.kH[0][0], 128, 32 * hd, 0, 0);
    v8f c = {};
    c = wmma_f16(a, b, c);
#pragma unroll
    for (int r = 0; r < 8; ++r)
      s.sc[hd][r + 8 * hi][ln15] = c[r] * SCALE_QK;
  }
  __syncthreads();

  // ---- softmax rows (mask cols >= SREAL for padded sequences)
  if (tid < 64) {
    const int hd = tid >> 4, row = tid & 15;
    float mx = -1e30f;
    for (int j = 0; j < SREAL; ++j) mx = fmaxf(mx, s.sc[hd][row][j]);
    float e[16], sum = 0.f;
    for (int j = 0; j < SREAL; ++j) {
      e[j] = __expf(s.sc[hd][row][j] - mx);
      sum += e[j];
    }
    const float inv = 1.f / sum;
#pragma unroll
    for (int j0 = 0; j0 < 32; j0 += 8) {
      v8h hv;
#pragma unroll
      for (int jj = 0; jj < 8; ++jj) {
        const int j = j0 + jj;
        hv[jj] = (j < SREAL) ? (_Float16)(e[j] * inv) : (_Float16)0.f;
      }
      *(v8h*)&s.attH[hd][row][j0] = hv;
    }
  }
  __syncthreads();

  // ---- O_h = A_h (16x32, zero-padded K) @ V_h (vT columns, contiguous)
  {
    const int hd = wave >> 1, half = wave & 1;
    const int col = 32 * hd + 16 * half + ln15;
    v16h a = load_a(&s.attH[hd][0][0], 32, 0);
    const _Float16* p = &s.vT[col][hi * 16];
    v16h b = join8(*(const v8h*)p, *(const v8h*)(p + 8));
    v8f c = {};
    c = wmma_f16(a, b, c);
#pragma unroll
    for (int r = 0; r < 8; ++r) s.oH[r + 8 * hi][col] = (_Float16)c[r];
  }
  __syncthreads();

  // ---- output projection + residual
  {
    v8f c = {};
#pragma unroll
    for (int k0 = 0; k0 < 128; k0 += 32) {
      v16h a = load_a(&s.oH[0][0], 128, k0);
      v16h b = load_bp(w.Wo, 8, k0 >> 5, wave);
      c = wmma_f16(a, b, c);
    }
    const float bias = w.bo[n0 + ln15];
#pragma unroll
    for (int r = 0; r < 8; ++r) s.xF[r + 8 * hi][n0 + ln15] += c[r] + bias;
  }
  __syncthreads();
  layernorm16(s.xF, w.g1, w.be1);
  __syncthreads();

  cast_act16(s);
  __syncthreads();

  // ---- FFN: stream F in 256-wide chunks, accumulate output in registers
  v8f acc = {};
  for (int ch = 0; ch < NCHUNK; ++ch) {
    if (lane == 0)
      __builtin_prefetch(w.W2 + (size_t)(ch * 256) * 128);  // global_prefetch_b8
#pragma unroll
    for (int t = 0; t < 2; ++t) {
      const int tt = 2 * wave + t;
      v8f c = {};
#pragma unroll
      for (int k0 = 0; k0 < 128; k0 += 32) {
        v16h a = load_a(&s.aH[0][0], 128, k0);
        v16h b = load_bp(w.W1, FDIM >> 4, k0 >> 5, ch * 16 + tt);
        c = wmma_f16(a, b, c);
      }
      const float bias = w.b1[ch * 256 + 16 * tt + ln15];
#pragma unroll
      for (int r = 0; r < 8; ++r)
        s.pH[r + 8 * hi][16 * tt + ln15] = (_Float16)fmaxf(0.f, c[r] + bias);
    }
    __syncthreads();
#pragma unroll
    for (int k0 = 0; k0 < 256; k0 += 32) {
      v16h a = load_a(&s.pH[0][0], 256, k0);
      v16h b = load_bp(w.W2, 8, ch * 8 + (k0 >> 5), wave);
      acc = wmma_f16(a, b, acc);
    }
    __syncthreads();
  }
  {
    const float bias = w.b2[n0 + ln15];
#pragma unroll
    for (int r = 0; r < 8; ++r) s.xF[r + 8 * hi][n0 + ln15] += acc[r] + bias;
  }
  __syncthreads();
  layernorm16(s.xF, w.g2, w.be2);
  __syncthreads();
}

// ------------------------------ pack kernels -------------------------------
// Repack W[K][N] (row-major f32) into WMMA B-fragment order:
// dst[((kt*NT+nt)*32 + lane)*16 + j] = W[kt*32 + ((lane>>4)<<4) + j][nt*16 + (lane&15)]
__global__ void pack_bfrag_kernel(const float* __restrict__ src,
                                  _Float16* __restrict__ dst, int K, int N) {
  const int i = blockIdx.x * 256 + threadIdx.x;
  if (i >= K * N) return;
  const int j = i & 15;
  const int lane = (i >> 4) & 31;
  const int tile = i >> 9;
  const int ntt = N >> 4;
  const int kt = tile / ntt, nt = tile - kt * ntt;
  const int k = kt * 32 + ((lane >> 4) << 4) + j;
  const int n = nt * 16 + (lane & 15);
  dst[i] = (_Float16)src[(size_t)k * N + n];
}

// conv2_W (128,16,3,3) -> fragment-order [K=160 (144 real, zero pad)][128]
__global__ void pack_conv2_kernel(const float* __restrict__ src,
                                  _Float16* __restrict__ dst) {
  const int i = blockIdx.x * 256 + threadIdx.x;
  if (i >= 160 * 128) return;
  const int j = i & 15;
  const int lane = (i >> 4) & 31;
  const int tile = i >> 9;
  const int kt = tile >> 3, nt = tile & 7;
  const int k = kt * 32 + ((lane >> 4) << 4) + j;  // K = c*9 + ky*3 + kx
  const int n = nt * 16 + (lane & 15);             // output channel
  dst[i] = (k < 144) ? (_Float16)src[(size_t)n * 144 + k] : (_Float16)0.f;
}

// ----------------------------- embed kernel --------------------------------
__global__ __launch_bounds__(128) void embed_kernel(
    const float* __restrict__ pos, const float* __restrict__ goal,
    const float* __restrict__ pW, const float* __restrict__ pb,
    const float* __restrict__ gW, const float* __restrict__ gb,
    float* __restrict__ h) {
  const int row = blockIdx.x, d = threadIdx.x;
  const float p0 = pos[row * 2], p1 = pos[row * 2 + 1];
  const float g0 = goal[row * 2] - p0, g1 = goal[row * 2 + 1] - p1;
  h[(size_t)row * ND + d] =
      p0 * pW[d] + p1 * pW[ND + d] + pb[d] + g0 * gW[d] + g1 * gW[ND + d] + gb[d];
}

// --------------------------- history encoder -------------------------------
__global__ __launch_bounds__(256) void hist_kernel(
    const float* __restrict__ history, TLayerW l0, TLayerW l1,
    float* __restrict__ hist_feat) {
  __shared__ SmemT s;
  const int tid = threadIdx.x;
  const int g = blockIdx.x;  // 0 .. B*T*N-1
  const float* hp = history + (size_t)g * NL * ND;
  for (int i = tid; i < NL * ND; i += 256) {
    const int pos = i >> 7, d = i & 127;
    const int j2 = d & ~1;  // arange(0,128,2)
    const float ang = (float)pos * __expf(-(float)j2 * (9.210340371976184f / 128.f));
    const float pe = (d & 1) ? __cosf(ang) : __sinf(ang);
    s.xF[0][i] = hp[i] + pe;
  }
  __syncthreads();
  tblock<16, 2048>(s, l0);
  tblock<16, 2048>(s, l1);
  if (tid < ND) {
    float sum = 0.f;
#pragma unroll
    for (int r = 0; r < NL; ++r) sum += s.xF[r][tid];
    hist_feat[(size_t)g * ND + tid] = sum * (1.f / (float)NL);
  }
}

// ------------------------ temporal transformer -----------------------------
__global__ __launch_bounds__(256) void temp_kernel(float* __restrict__ h,
                                                   TLayerW l0, TLayerW l1) {
  __shared__ SmemT s;
  const int tid = threadIdx.x;
  const int g = blockIdx.x;  // b*N + n
  const int b = g >> 6, n = g & 63;
  for (int i = tid; i < 16 * ND; i += 256) {
    const int t = i >> 7, d = i & 127;
    s.xF[t][d] = (t < NT) ? h[((((size_t)b * NT + t) * NN) + n) * ND + d] : 0.f;
  }
  __syncthreads();
  tblock<8, 512>(s, l0);
  tblock<8, 512>(s, l1);
  for (int i = tid; i < NT * ND; i += 256) {
    const int t = i >> 7, d = i & 127;
    h[((((size_t)b * NT + t) * NN) + n) * ND + d] = s.xF[t][d];
  }
}

// -------------------------- local map CNN ----------------------------------
struct SmemC {                  // 46,848 B
  float    mapL[32][32];
  float    w1L[16][9];
  float    b1L[16];
  _Float16 x1H[16][34][34];     // conv1 output, zero-padded border
  _Float16 aH[16][160];         // im2col pixel tile (K padded 144->160)
};

__global__ __launch_bounds__(256) void cnn_kernel(
    const float* __restrict__ maps, const float* __restrict__ cw1,
    const float* __restrict__ cb1, const _Float16* __restrict__ W2p,
    const float* __restrict__ cb2, float* __restrict__ map_feat) {
  __shared__ SmemC s;
  const int tid = threadIdx.x;
  const int wave = tid >> 5, lane = tid & 31;
  const int ln15 = lane & 15, hi = lane >> 4;
  const int g = blockIdx.x;
  const float* mp = maps + (size_t)g * 1024;

  for (int i = tid; i < 1024; i += 256) s.mapL[0][i] = mp[i];
  if (tid < 144) s.w1L[0][tid] = cw1[tid];
  if (tid < 16) s.b1L[tid] = cb1[tid];
  for (int i = tid; i < 16 * 34 * 34; i += 256)
    (&s.x1H[0][0][0])[i] = (_Float16)0.f;
  __syncthreads();

  // conv1 (1->16ch, 3x3 SAME) + relu into padded f16 tensor
  for (int p = tid; p < 1024; p += 256) {
    const int y = p >> 5, x = p & 31;
    float win[9];
#pragma unroll
    for (int ky = 0; ky < 3; ++ky)
#pragma unroll
      for (int kx = 0; kx < 3; ++kx) {
        const int yy = y + ky - 1, xx = x + kx - 1;
        win[ky * 3 + kx] =
            (yy >= 0 && yy < 32 && xx >= 0 && xx < 32) ? s.mapL[yy][xx] : 0.f;
      }
#pragma unroll
    for (int c = 0; c < 16; ++c) {
      float acc = s.b1L[c];
#pragma unroll
      for (int k = 0; k < 9; ++k) acc += s.w1L[c][k] * win[k];
      s.x1H[c][y + 1][x + 1] = (_Float16)fmaxf(0.f, acc);
    }
  }
  __syncthreads();

  // conv2 implicit GEMM: 64 pixel-tiles of 16, each wave one 16-channel tile.
  float accs[8] = {0.f, 0.f, 0.f, 0.f, 0.f, 0.f, 0.f, 0.f};
  for (int pt = 0; pt < 64; ++pt) {
    // im2col tile build: 320 8-half segments, one b128 store each
    for (int seg = tid; seg < 320; seg += 256) {
      const int row = seg / 20, k8 = (seg % 20) * 8;
      v8h hv;
#pragma unroll
      for (int jj = 0; jj < 8; ++jj) {
        const int kk = k8 + jj;
        _Float16 v = (_Float16)0.f;
        if (kk < 144) {
          const int c = kk / 9, r9 = kk % 9;
          const int ky = r9 / 3, kx = r9 % 3;
          const int px = pt * 16 + row;
          const int y = px >> 5, x = px & 31;
          v = s.x1H[c][y + ky][x + kx];
        }
        hv[jj] = v;
      }
      *(v8h*)&s.aH[row][k8] = hv;
    }
    __syncthreads();
    v8f c = {};
#pragma unroll
    for (int k0 = 0; k0 < 160; k0 += 32) {
      v16h a = load_a(&s.aH[0][0], 160, k0);
      v16h b = load_bp(W2p, 8, k0 >> 5, wave);
      c = wmma_f16(a, b, c);
    }
    const float bias = cb2[wave * 16 + ln15];
#pragma unroll
    for (int r = 0; r < 8; ++r) accs[r] += fmaxf(0.f, c[r] + bias);
    __syncthreads();
  }
  float tot = 0.f;
#pragma unroll
  for (int r = 0; r < 8; ++r) tot += accs[r];
  tot += __shfl_xor(tot, 16);  // combine the two M-halves
  if (lane < 16)
    map_feat[(size_t)g * ND + wave * 16 + lane] = tot * (1.f / 1024.f);
  (void)hi;
}

// --------------------------- feature combine -------------------------------
__global__ __launch_bounds__(256) void addfeat_kernel(
    float* __restrict__ h, const float* __restrict__ hf,
    const float* __restrict__ mf) {
  const int i = blockIdx.x * 256 + threadIdx.x;
  if (i < 2048 * ND) h[i] += hf[i] + mf[i];
}

// --------------------- spatial graph-attention layer -----------------------
struct SmemS {                  // 60,544 B
  _Float16 xH[64][128];
  _Float16 kH[64][128];
  _Float16 vT[128][64];         // V transposed: vT[d][token]
  _Float16 qoH[16][128];        // Q row-tile; reused in-place for O
  float    sc[16][64];
  _Float16 aW[16][64];
  float    red[9][16][2];
};

__global__ __launch_bounds__(256) void spat_kernel(
    float* __restrict__ h, const float* __restrict__ adj,
    const float* __restrict__ ef, SLayerW w) {
  __shared__ SmemS s;
  const int tid = threadIdx.x;
  const int wave = tid >> 5, lane = tid & 31;
  const int ln15 = lane & 15, hi = lane >> 4;
  const int bt = blockIdx.x;  // b*T + t
  float* hx = h + (size_t)bt * NN * ND;
  const float* adjp = adj + (size_t)bt * NN * NN;
  const float* efp = ef + (size_t)bt * NN * NN * NE;

  for (int i0 = tid * 8; i0 < NN * ND; i0 += 2048) {
    v8h hv;
#pragma unroll
    for (int j = 0; j < 8; ++j) hv[j] = (_Float16)hx[i0 + j];
    *(v8h*)(&s.xH[0][0] + i0) = hv;
  }
  __syncthreads();

  // K (row-major) and V (transposed) for all 64 tokens: 64 tiles / 8 waves
  for (int tt = wave; tt < 64; tt += 8) {
    const int mat = tt >> 5;  // 0=K 1=V
    const int t2 = tt & 31;
    const int rt = t2 >> 3, ct = t2 & 7;
    const _Float16* Wm = mat ? w.Wv : w.Wk;
    const float* bm = mat ? w.bv : w.bk;
    v8f c = {};
#pragma unroll
    for (int k0 = 0; k0 < 128; k0 += 32) {
      v16h a = load_a(&s.xH[rt * 16][0], 128, k0);
      v16h b = load_bp(Wm, 8, k0 >> 5, ct);
      c = wmma_f16(a, b, c);
    }
    const float bias = bm[ct * 16 + ln15];
    if (mat == 0) {
#pragma unroll
      for (int r = 0; r < 8; ++r)
        s.kH[rt * 16 + r + 8 * hi][ct * 16 + ln15] = (_Float16)(c[r] + bias);
    } else {
      v8h hv;
#pragma unroll
      for (int r = 0; r < 8; ++r) hv[r] = (_Float16)(c[r] + bias);
      *(v8h*)&s.vT[ct * 16 + ln15][rt * 16 + 8 * hi] = hv;
    }
  }
  __syncthreads();

  for (int rt = 0; rt < 4; ++rt) {
    const int row0 = rt * 16;
    // Q row-tile
    {
      v8f c = {};
#pragma unroll
      for (int k0 = 0; k0 < 128; k0 += 32) {
        v16h a = load_a(&s.xH[row0][0], 128, k0);
        v16h b = load_bp(w.Wq, 8, k0 >> 5, wave);
        c = wmma_f16(a, b, c);
      }
      const float bias = w.bq[wave * 16 + ln15];
#pragma unroll
      for (int r = 0; r < 8; ++r)
        s.qoH[r + 8 * hi][wave * 16 + ln15] = (_Float16)(c[r] + bias);
    }
    __syncthreads();

    for (int hd = 0; hd < 4; ++hd) {
      // scores: 4 column-tiles over 64 keys
      if (wave < 4) {
        const int ct = wave;
        v16h a = load_a(&s.qoH[0][0], 128, 32 * hd);
        v16h b = load_bT(&s.kH[ct * 16][0], 128, 32 * hd, 0, 0);
        v8f c = {};
        c = wmma_f16(a, b, c);
#pragma unroll
        for (int r = 0; r < 8; ++r)
          s.sc[r + 8 * hi][ct * 16 + ln15] = c[r] * SCALE_QK;
      }
      __syncthreads();
      // edge-feature bias + adjacency mask
      for (int e4 = tid; e4 < 16 * 64; e4 += 256) {
        const int row = e4 >> 6, kc = e4 & 63;
        const int q = row0 + row;
        const float4* ev = (const float4*)(efp + ((size_t)q * NN + kc) * NE);
        float bsum = w.be[hd];
#pragma unroll
        for (int e4i = 0; e4i < 4; ++e4i) {
          const float4 v = ev[e4i];
          bsum += v.x * w.We[(e4i * 4 + 0) * 4 + hd] +
                  v.y * w.We[(e4i * 4 + 1) * 4 + hd] +
                  v.z * w.We[(e4i * 4 + 2) * 4 + hd] +
                  v.w * w.We[(e4i * 4 + 3) * 4 + hd];
        }
        float val = s.sc[row][kc] + bsum;
        if (adjp[q * NN + kc] == 0.f) val = -1e9f;
        s.sc[row][kc] = val;
      }
      __syncthreads();
      if (tid < 16) {
        const int row = tid;
        float mx = -1e30f;
        for (int j = 0; j < 64; ++j) mx = fmaxf(mx, s.sc[row][j]);
        float sum = 0.f;
        for (int j = 0; j < 64; ++j) sum += __expf(s.sc[row][j] - mx);
        const float inv = 1.f / sum;
        for (int j0 = 0; j0 < 64; j0 += 8) {
          v8h hv;
#pragma unroll
          for (int jj = 0; jj < 8; ++jj)
            hv[jj] = (_Float16)(__expf(s.sc[row][j0 + jj] - mx) * inv);
          *(v8h*)&s.aW[row][j0] = hv;
        }
      }
      __syncthreads();
      // O_h: overwrite the (already consumed) Q columns of this head
      if (wave < 2) {
        const int half = wave;
        const int col = 32 * hd + 16 * half + ln15;
        v8f c = {};
#pragma unroll
        for (int k0 = 0; k0 < 64; k0 += 32) {
          v16h a = load_a(&s.aW[0][0], 64, k0);
          const _Float16* p = &s.vT[col][k0 + hi * 16];
          v16h b = join8(*(const v8h*)p, *(const v8h*)(p + 8));
          c = wmma_f16(a, b, c);
        }
#pragma unroll
        for (int r = 0; r < 8; ++r) s.qoH[r + 8 * hi][col] = (_Float16)c[r];
      }
      __syncthreads();
    }

    // output projection + residual + LN (cross-wave row stats via LDS)
    float vals[8];
    {
      v8f c = {};
#pragma unroll
      for (int k0 = 0; k0 < 128; k0 += 32) {
        v16h a = load_a(&s.qoH[0][0], 128, k0);
        v16h b = load_bp(w.Wo, 8, k0 >> 5, wave);
        c = wmma_f16(a, b, c);
      }
      const float bias = w.bo[wave * 16 + ln15];
#pragma unroll
      for (int r = 0; r < 8; ++r) {
        const int m = r + 8 * hi;
        vals[r] = c[r] + bias + hx[(size_t)(row0 + m) * ND + wave * 16 + ln15];
      }
    }
#pragma unroll
    for (int r = 0; r < 8; ++r) {
      float sv = vals[r], sq = vals[r] * vals[r];
#pragma unroll
      for (int m2 = 8; m2 >= 1; m2 >>= 1) {
        sv += __shfl_xor(sv, m2);
        sq += __shfl_xor(sq, m2);
      }
      if (ln15 == 0) {
        s.red[wave][r + 8 * hi][0] = sv;
        s.red[wave][r + 8 * hi][1] = sq;
      }
    }
    __syncthreads();
    if (tid < 16) {
      float ts = 0.f, tq = 0.f;
      for (int w8 = 0; w8 < 8; ++w8) {
        ts += s.red[w8][tid][0];
        tq += s.red[w8][tid][1];
      }
      const float mean = ts * (1.f / 128.f);
      const float var = tq * (1.f / 128.f) - mean * mean;
      s.red[8][tid][0] = mean;
      s.red[8][tid][1] = rsqrtf(var + 1e-5f);
    }
    __syncthreads();
#pragma unroll
    for (int r = 0; r < 8; ++r) {
      const int m = r + 8 * hi;
      const int col = wave * 16 + ln15;
      const float o =
          (vals[r] - s.red[8][m][0]) * s.red[8][m][1] * w.g[col] + w.bln[col];
      hx[(size_t)(row0 + m) * ND + col] = o;
    }
    __syncthreads();
  }
}

// ----------------------------- output head ---------------------------------
__global__ __launch_bounds__(256) void out_kernel(const float* __restrict__ h,
                                                  const float* __restrict__ oW,
                                                  const float* __restrict__ ob,
                                                  float* __restrict__ out) {
  const int t = threadIdx.x;  // 256 = B*N
  const int b = t >> 6, n = t & 63;
  const float* x = h + (((size_t)b * NT + (NT - 1)) * NN + n) * ND;
  float s0 = ob[0], s1 = ob[1];
  for (int d = 0; d < ND; ++d) {
    s0 += x[d] * oW[2 * d];
    s1 += x[d] * oW[2 * d + 1];
  }
  for (int p = 0; p < NP; ++p) {
    float* o = out + (((size_t)b * NP + p) * NN + n) * NOUT;
    o[0] = s0;
    o[1] = s1;
  }
}

// ------------------------------- launcher ----------------------------------
extern "C" void kernel_launch(void* const* d_in, const int* in_sizes, int n_in,
                              void* d_out, int out_size, void* d_ws,
                              size_t ws_size, hipStream_t stream) {
  (void)in_sizes; (void)n_in; (void)out_size; (void)ws_size;
  auto F32 = [&](int i) { return (const float*)d_in[i]; };

  const float* positions = F32(0);
  const float* goals = F32(1);
  const float* history = F32(2);
  const float* local_maps = F32(3);
  const float* adjacency = F32(4);
  const float* edge_features = F32(5);

  // workspace carve-up
  float* h = (float*)d_ws;                 // (B*T*N, D)
  float* hist_feat = h + 2048 * ND;
  float* map_feat = hist_feat + 2048 * ND;
  _Float16* fp = (_Float16*)(map_feat + 2048 * ND);

  auto pack = [&](int idx, int K, int N) -> const _Float16* {
    _Float16* dst = fp;
    fp += (size_t)K * N;
    pack_bfrag_kernel<<<((K * N) + 255) / 256, 256, 0, stream>>>(F32(idx), dst,
                                                                 K, N);
    return dst;
  };
  auto tl = [&](int base, int F) {
    TLayerW w;
    w.Wq = pack(base + 0, 128, 128);
    w.Wk = pack(base + 1, 128, 128);
    w.Wv = pack(base + 2, 128, 128);
    w.Wo = pack(base + 3, 128, 128);
    w.bq = F32(base + 4); w.bk = F32(base + 5);
    w.bv = F32(base + 6); w.bo = F32(base + 7);
    w.g1 = F32(base + 8); w.be1 = F32(base + 9);
    w.W1 = pack(base + 10, 128, F);
    w.b1 = F32(base + 11);
    w.W2 = pack(base + 12, F, 128);
    w.b2 = F32(base + 13);
    w.g2 = F32(base + 14); w.be2 = F32(base + 15);
    return w;
  };
  auto sl = [&](int base) {
    SLayerW w;
    w.Wq = pack(base + 0, 128, 128);
    w.Wk = pack(base + 1, 128, 128);
    w.Wv = pack(base + 2, 128, 128);
    w.Wo = pack(base + 3, 128, 128);
    w.bq = F32(base + 4); w.bk = F32(base + 5);
    w.bv = F32(base + 6); w.bo = F32(base + 7);
    w.We = F32(base + 8); w.be = F32(base + 9);
    w.g = F32(base + 10); w.bln = F32(base + 11);
    return w;
  };

  TLayerW hl0 = tl(10, 2048), hl1 = tl(26, 2048);
  TLayerW tl0 = tl(46, 512), tl1 = tl(62, 512);
  SLayerW sp0 = sl(78), sp1 = sl(90);
  _Float16* W2p = fp;
  fp += 160 * 128;
  pack_conv2_kernel<<<80, 256, 0, stream>>>(F32(44), W2p);

  embed_kernel<<<2048, 128, 0, stream>>>(positions, goals, F32(6), F32(7),
                                         F32(8), F32(9), h);
  hist_kernel<<<2048, 256, 0, stream>>>(history, hl0, hl1, hist_feat);
  cnn_kernel<<<2048, 256, 0, stream>>>(local_maps, F32(42), F32(43), W2p,
                                       F32(45), map_feat);
  addfeat_kernel<<<1024, 256, 0, stream>>>(h, hist_feat, map_feat);
  temp_kernel<<<256, 256, 0, stream>>>(h, tl0, tl1);
  spat_kernel<<<32, 256, 0, stream>>>(h, adjacency, edge_features, sp0);
  spat_kernel<<<32, 256, 0, stream>>>(h, adjacency, edge_features, sp1);
  out_kernel<<<1, 256, 0, stream>>>(h, F32(102), F32(103), (float*)d_out);
}
